// Feast_GCN_16028817949352
// MI455X (gfx1250) — compile-verified, matrix operand order checked by reading, not run on hardware
//
#include <hip/hip_runtime.h>
#include <hip/hip_bf16.h>

// FeaStNet GCN for MI455X (gfx1250, wave32).
// - Edge scatter kept L2-resident (agg = 153.6 MB < 192 MB L2).
// - Dense head-transforms: 24 back-to-back v_wmma_f32_16x16x32_bf16 per
//   output tile; W pre-packed to bf16 in fragment layout once per layer so
//   the GEMM hot loop has no conversion VALU. 1/deg folded into epilogue.

#define NN 50000
#define EE 800000
#define DD 128
#define HH 6

typedef __attribute__((ext_vector_type(16))) __bf16 v16bf;
typedef __attribute__((ext_vector_type(8)))  float  v8f;

// ---------------------------------------------------------------- utilities

__global__ void zero_kernel(float* __restrict__ p, long n) {
    long i = (long)blockIdx.x * blockDim.x + threadIdx.x;
    long stride = (long)gridDim.x * blockDim.x;
    for (; i < n; i += stride) p[i] = 0.0f;
}

__global__ void degree_kernel(const int* __restrict__ dst, float* __restrict__ deg, int e) {
    int i = blockIdx.x * blockDim.x + threadIdx.x;
    if (i < e) atomicAdd(&deg[dst[i]], 1.0f);
}

__global__ void invdeg_kernel(const float* __restrict__ deg, float* __restrict__ invdeg, int n) {
    int i = blockIdx.x * blockDim.x + threadIdx.x;
    if (i < n) invdeg[i] = 1.0f / fmaxf(deg[i], 1.0f);
}

// ------------------------------------------------------- encoder (6 -> 128)

__global__ void encoder_kernel(const float* __restrict__ pos, const float* __restrict__ nrm,
                               const float* __restrict__ w1, const float* __restrict__ b1,
                               float* __restrict__ x, int n) {
    int t = blockIdx.x * blockDim.x + threadIdx.x;
    int node = t >> 7, dout = t & 127;
    if (node >= n) return;
    const float* w = w1 + dout * 6;
    float v = b1[dout];
    v += pos[node * 3 + 0] * w[0] + pos[node * 3 + 1] * w[1] + pos[node * 3 + 2] * w[2];
    v += nrm[node * 3 + 0] * w[3] + nrm[node * 3 + 1] * w[4] + nrm[node * 3 + 2] * w[5];
    x[(size_t)node * DD + dout] = fmaxf(v, 0.0f);
}

// ------------------------------------------- edge phase: softmax + scatter
// One wave32 per edge; lane l owns features [4l, 4l+4).

__global__ void __launch_bounds__(256)
edge_kernel(const float* __restrict__ x, const int* __restrict__ src,
            const int* __restrict__ dst, const float* __restrict__ u,
            const float* __restrict__ c, float* __restrict__ agg, int e) {
    int lane = threadIdx.x & 31;
    int edge = (int)((blockIdx.x * (unsigned)blockDim.x + threadIdx.x) >> 5);
    if (edge >= e) return;
    int s = src[edge], d = dst[edge];

    float4 xj = reinterpret_cast<const float4*>(x + (size_t)s * DD)[lane];
    float4 xi = reinterpret_cast<const float4*>(x + (size_t)d * DD)[lane];
    float d0 = xj.x - xi.x, d1 = xj.y - xi.y, d2 = xj.z - xi.z, d3 = xj.w - xi.w;

    float logit[HH];
#pragma unroll
    for (int h = 0; h < HH; ++h) {
        float4 u4 = reinterpret_cast<const float4*>(u + h * DD)[lane];
        float p = d0 * u4.x + d1 * u4.y + d2 * u4.z + d3 * u4.w;
#pragma unroll
        for (int off = 16; off > 0; off >>= 1) p += __shfl_xor(p, off, 32);
        logit[h] = p + c[h];
    }
    float m = logit[0];
#pragma unroll
    for (int h = 1; h < HH; ++h) m = fmaxf(m, logit[h]);
    float q[HH], sum = 0.0f;
#pragma unroll
    for (int h = 0; h < HH; ++h) { q[h] = __expf(logit[h] - m); sum += q[h]; }
    float inv = 1.0f / sum;

    int f = lane * 4;
#pragma unroll
    for (int h = 0; h < HH; ++h) {
        float w = q[h] * inv;
        float* a = agg + ((size_t)h * NN + d) * DD + f;
        atomicAdd(a + 0, w * xj.x);
        atomicAdd(a + 1, w * xj.y);
        atomicAdd(a + 2, w * xj.z);
        atomicAdd(a + 3, w * xj.w);
    }
}

// ---------------------------- W pre-pack: f32 -> bf16 in B-fragment layout
// Wp[(((h*4 + kt)*2 + half)*128 + col)*16 + i] where, per ISA 7.12.2:
//   i in [0,8):  K = kt*32 + half*8 + i
//   i in [8,16): K = kt*32 + 16 + half*8 + (i-8)

__global__ void packW_kernel(const float* __restrict__ Wg, __bf16* __restrict__ Wp) {
    int t = blockIdx.x * blockDim.x + threadIdx.x;   // H*4*2*128 = 6144
    if (t >= HH * 4 * 2 * DD) return;
    int col = t & 127;
    int s   = (t >> 7) & 1;
    int kt  = (t >> 8) & 3;
    int h   = t >> 10;
    const float* wrow = Wg + ((size_t)h * DD + col) * DD;
    __bf16* out = Wp + (size_t)t * 16;
#pragma unroll
    for (int i = 0; i < 8; ++i) {
        out[i]     = (__bf16)wrow[kt * 32 + s * 8 + i];
        out[8 + i] = (__bf16)wrow[kt * 32 + 16 + s * 8 + i];
    }
}

// --------------------------------- node phase: out = relu(bg + id*Σ_h A_h W_h^T)
// Block = 256 threads (8 waves): 16-node row tile; wave w owns output cols
// [16w, 16w+16). All 6 heads staged in LDS at once -> one barrier pair and
// 24 back-to-back v_wmma_f32_16x16x32_bf16 per tile. invdeg applied in the
// epilogue (it is per-row, so it commutes with the K/head reduction).

#define LDA (DD + 8)   // pad LDS rows to dodge 64-bank conflicts

__global__ void __launch_bounds__(256)
feast_gemm_kernel(const float* __restrict__ agg, const float* __restrict__ invdeg,
                  const __bf16* __restrict__ Wp, const float* __restrict__ bg,
                  float* __restrict__ xout) {
    __shared__ __bf16 As[HH][16 * LDA];   // ~26 KB of 320 KB/WGP

    const int tid  = threadIdx.x;
    const int lane = tid & 31;
    const int wave = tid >> 5;         // 0..7 -> output column tile
    const int node0 = blockIdx.x * 16; // N = 50000 = 16 * 3125 exactly
    const int m    = lane & 15;        // row (A) / col (B,C,D) within tile
    const int half = lane >> 4;        // K-half selector per wave32 layout
    const int col  = wave * 16 + m;    // global output column

    // Stage raw A tiles (6 x 16 x 128) into LDS as bf16 (no scaling here).
    {
        const int r     = tid >> 4;        // row 0..15 owned by this thread
        const int cbase = (tid & 15) * 8;  // 8-col slice
#pragma unroll
        for (int h = 0; h < HH; ++h) {
            const float* srcp = agg + ((size_t)h * NN + node0 + r) * DD + cbase;
#pragma unroll
            for (int j = 0; j < 8; ++j)
                As[h][r * LDA + cbase + j] = (__bf16)srcp[j];
        }
    }
    __builtin_prefetch(Wp + (size_t)half * DD * 16 + (size_t)col * 16, 0, 0); // global_prefetch_b8
    __syncthreads();

    v8f acc = {};
#pragma unroll
    for (int h = 0; h < HH; ++h) {
#pragma unroll
        for (int kt = 0; kt < 4; ++kt) {
            // B fragment: one aligned 32-byte load, already in fragment order.
            v16bf b = *reinterpret_cast<const v16bf*>(
                Wp + ((size_t)(((h * 4 + kt) * 2 + half) * DD) + col) * 16);
            // A fragment: two contiguous 16-byte LDS reads.
            const __bf16* ap = &As[h][m * LDA + kt * 32 + half * 8];
            v16bf a;
#pragma unroll
            for (int i = 0; i < 8; ++i) { a[i] = ap[i]; a[8 + i] = ap[16 + i]; }
            acc = __builtin_amdgcn_wmma_f32_16x16x32_bf16(
                false, a, false, b, (short)0, acc, false, false);
        }
    }

    // Epilogue: C/D layout -> row = r + 8*half, col = lane&15.
    // out = relu(invdeg[row] * acc + bias)
    const float bias = bg[col];
#pragma unroll
    for (int r = 0; r < 8; ++r) {
        int node = node0 + r + half * 8;
        xout[(size_t)node * DD + col] = fmaxf(acc[r] * invdeg[node] + bias, 0.0f);
    }
}

// ------------------------------------------------------- decoder (128 -> 3)
// One wave32 per node: coalesced float4 row load, shfl-reduced dots.

__global__ void __launch_bounds__(256)
decoder_kernel(const float* __restrict__ x, const float* __restrict__ w2,
               const float* __restrict__ b2, float* __restrict__ out, int n) {
    int lane = threadIdx.x & 31;
    int node = (int)((blockIdx.x * (unsigned)blockDim.x + threadIdx.x) >> 5);
    if (node >= n) return;
    float4 xv = reinterpret_cast<const float4*>(x + (size_t)node * DD)[lane];
    float p0 = 0.0f, p1 = 0.0f, p2 = 0.0f;
    {
        float4 wv = reinterpret_cast<const float4*>(w2 + 0 * DD)[lane];
        p0 = xv.x * wv.x + xv.y * wv.y + xv.z * wv.z + xv.w * wv.w;
        wv = reinterpret_cast<const float4*>(w2 + 1 * DD)[lane];
        p1 = xv.x * wv.x + xv.y * wv.y + xv.z * wv.z + xv.w * wv.w;
        wv = reinterpret_cast<const float4*>(w2 + 2 * DD)[lane];
        p2 = xv.x * wv.x + xv.y * wv.y + xv.z * wv.z + xv.w * wv.w;
    }
#pragma unroll
    for (int off = 16; off > 0; off >>= 1) {
        p0 += __shfl_xor(p0, off, 32);
        p1 += __shfl_xor(p1, off, 32);
        p2 += __shfl_xor(p2, off, 32);
    }
    if (lane < 3) {
        float r = (lane == 0) ? p0 : ((lane == 1) ? p1 : p2);
        out[(size_t)node * 3 + lane] = r + b2[lane];
    }
}

// ----------------------------------------------------------------- launcher

extern "C" void kernel_launch(void* const* d_in, const int* in_sizes, int n_in,
                              void* d_out, int out_size, void* d_ws, size_t ws_size,
                              hipStream_t stream) {
    (void)in_sizes; (void)n_in; (void)out_size; (void)ws_size;

    const float* pos = (const float*)d_in[0];
    const float* nrm = (const float*)d_in[1];
    const int*   ei  = (const int*)d_in[2];
    const int*   src = ei;
    const int*   dst = ei + EE;
    const float* w1  = (const float*)d_in[3];
    const float* b1  = (const float*)d_in[4];
    const float* w2  = (const float*)d_in[5];
    const float* b2  = (const float*)d_in[6];
    const float *Wg[4], *u[4], *cc[4], *bg[4];
    for (int l = 0; l < 4; ++l) {
        Wg[l] = (const float*)d_in[7 + 4 * l];
        u[l]  = (const float*)d_in[8 + 4 * l];
        cc[l] = (const float*)d_in[9 + 4 * l];
        bg[l] = (const float*)d_in[10 + 4 * l];
    }

    // Workspace layout: xA | xB | deg | invdeg | agg[H][N][D] | Wpack(bf16)
    float* ws     = (float*)d_ws;
    float* xA     = ws;
    float* xB     = xA + (size_t)NN * DD;
    float* deg    = xB + (size_t)NN * DD;
    float* invdeg = deg + NN;
    float* agg    = invdeg + NN;
    __bf16* Wp    = (__bf16*)(agg + (size_t)HH * NN * DD);  // 32B-aligned offset

    zero_kernel<<<1024, 256, 0, stream>>>(deg, NN);
    degree_kernel<<<(EE + 255) / 256, 256, 0, stream>>>(dst, deg, EE);
    invdeg_kernel<<<(NN + 255) / 256, 256, 0, stream>>>(deg, invdeg, NN);
    encoder_kernel<<<(NN * DD + 255) / 256, 256, 0, stream>>>(pos, nrm, w1, b1, xA, NN);

    float* xc = xA;
    float* xn = xB;
    for (int l = 0; l < 4; ++l) {
        zero_kernel<<<8192, 256, 0, stream>>>(agg, (long)HH * NN * DD);
        packW_kernel<<<(HH * 4 * 2 * DD + 255) / 256, 256, 0, stream>>>(Wg[l], Wp);
        edge_kernel<<<(EE + 7) / 8, 256, 0, stream>>>(xc, src, dst, u[l], cc[l], agg, EE);
        feast_gemm_kernel<<<NN / 16, 256, 0, stream>>>(agg, invdeg, Wp, bg[l], xn);
        float* t = xc; xc = xn; xn = t;
    }

    decoder_kernel<<<(NN * 32 + 255) / 256, 256, 0, stream>>>(xc, w2, b2, (float*)d_out, NN);
}
